// GATNodeRegression_88613765251904
// MI455X (gfx1250) — compile-verified, matrix operand order checked by reading, not run on hardware
//
#include <hip/hip_runtime.h>
#include <hip/hip_bf16.h>

typedef __attribute__((ext_vector_type(16))) _Float16 v16h;
typedef __attribute__((ext_vector_type(8)))  float    v8f;

#define HID 64
#define HEADS 4
#define OUTC 16
#define NEG_SLOPE 0.2f

// ---------------------------------------------------------------------------
// Zero fill
// ---------------------------------------------------------------------------
__global__ void zero_f32(float* __restrict__ p, size_t n) {
    size_t t = (size_t)blockIdx.x * blockDim.x + threadIdx.x;
    if (t < n) p[t] = 0.0f;
}

// ---------------------------------------------------------------------------
// Degree: deg[dst] += 1 per edge (edges identical across layers -> once)
// ---------------------------------------------------------------------------
__global__ void gat_degree(const long long* __restrict__ dst,
                           float* __restrict__ deg, long long nedges) {
    long long t = (long long)blockIdx.x * blockDim.x + threadIdx.x;
    if (t < nedges) unsafeAtomicAdd(&deg[(int)dst[t]], 1.0f);
}

// In-place reciprocal (unused nodes -> inf, never read by edge kernel)
__global__ void recip_f32(float* __restrict__ p, int n) {
    int t = blockIdx.x * blockDim.x + threadIdx.x;
    if (t < n) p[t] = 1.0f / p[t];
}

// ---------------------------------------------------------------------------
// WMMA GEMM: H[n, 0:64] = X[n, 0:cin] @ W^T + bias
// One wave per 16(M) x 16(N) tile; K in steps of 32 via
// v_wmma_f32_16x16x32_f16 (f16 inputs, f32 accumulate).
//
// 16-bit A 16x32 layout: lane l holds row M=l%16 with K chunks
// {kh..kh+7, kh+16..kh+23}, kh = 8*(lane>=16) -> two contiguous float4 pairs,
// loaded as global_load_b128. W row-major [n][k] gives B (=W^T) the identical
// per-lane gather with row -> output channel.
// ---------------------------------------------------------------------------
__global__ __launch_bounds__(256) void gat_gemm(
        const float* __restrict__ X, const float* __restrict__ W,
        const float* __restrict__ bias, float* __restrict__ H,
        int nnodes, int cin) {
    int wave = threadIdx.x >> 5;
    int lane = threadIdx.x & 31;
    int tile = blockIdx.x * 8 + wave;
    int ntileM = (nnodes + 15) >> 4;
    if (tile >= ntileM * 4) return;           // wave-uniform exit

    int rt = tile >> 2;                       // M tile
    int ct = tile & 3;                        // N tile (== head)
    int r  = lane & 15;
    int kh = (lane >> 4) * 8;                 // K half select: 0 or 8

    int mrow = rt * 16 + r; if (mrow >= nnodes) mrow = nnodes - 1;
    int ncol = ct * 16 + r;                   // output channel for this lane

    const float* xrow = X + (size_t)mrow * cin;
    const float* wrow = W + (size_t)ncol * cin;

    v8f c = {};
    for (int k0 = 0; k0 < cin; k0 += 32) {
        const float4* xq = (const float4*)(xrow + k0 + kh);
        const float4* wq = (const float4*)(wrow + k0 + kh);
        float4 x0 = xq[0], x1 = xq[1];        // K = kh .. kh+7
        float4 x2 = xq[4], x3 = xq[5];        // K = kh+16 .. kh+23
        float4 w0 = wq[0], w1 = wq[1];
        float4 w2 = wq[4], w3 = wq[5];

        v16h a, b;
        a[0]=(_Float16)x0.x; a[1]=(_Float16)x0.y; a[2]=(_Float16)x0.z; a[3]=(_Float16)x0.w;
        a[4]=(_Float16)x1.x; a[5]=(_Float16)x1.y; a[6]=(_Float16)x1.z; a[7]=(_Float16)x1.w;
        a[8]=(_Float16)x2.x; a[9]=(_Float16)x2.y; a[10]=(_Float16)x2.z; a[11]=(_Float16)x2.w;
        a[12]=(_Float16)x3.x; a[13]=(_Float16)x3.y; a[14]=(_Float16)x3.z; a[15]=(_Float16)x3.w;
        b[0]=(_Float16)w0.x; b[1]=(_Float16)w0.y; b[2]=(_Float16)w0.z; b[3]=(_Float16)w0.w;
        b[4]=(_Float16)w1.x; b[5]=(_Float16)w1.y; b[6]=(_Float16)w1.z; b[7]=(_Float16)w1.w;
        b[8]=(_Float16)w2.x; b[9]=(_Float16)w2.y; b[10]=(_Float16)w2.z; b[11]=(_Float16)w2.w;
        b[12]=(_Float16)w3.x; b[13]=(_Float16)w3.y; b[14]=(_Float16)w3.z; b[15]=(_Float16)w3.w;

        c = __builtin_amdgcn_wmma_f32_16x16x32_f16(
                /*neg_a=*/false, a, /*neg_b=*/false, b,
                /*c_mod=*/(short)0, c, /*reuse_a=*/false, /*reuse_b=*/false);
    }

    // C/D layout: VGPR v -> row rt*16 + v + 8*(lane>=16), col = ct*16 + lane%16
    float bv = bias[ncol];
    int mbase = rt * 16 + (lane >> 4) * 8;
    if (mbase + 8 <= nnodes) {                // full tile: unconditional stores
#pragma unroll
        for (int v = 0; v < 8; ++v)
            H[(size_t)(mbase + v) * HID + ncol] = c[v] + bv;
    } else {
#pragma unroll
        for (int v = 0; v < 8; ++v)
            if (mbase + v < nnodes)
                H[(size_t)(mbase + v) * HID + ncol] = c[v] + bv;
    }
}

// ---------------------------------------------------------------------------
// Per-node attention scores: si[n,h] = h[n,h,:] . att[h,0:16]
//                            sj[n,h] = h[n,h,:] . att[h,16:32]
// ---------------------------------------------------------------------------
__global__ void gat_scores(const float* __restrict__ H,
                           const float* __restrict__ att,
                           float* __restrict__ si, float* __restrict__ sj,
                           int nnodes) {
    int t = blockIdx.x * blockDim.x + threadIdx.x;
    if (t >= nnodes * HEADS) return;
    int n = t >> 2, h = t & 3;
    const float* hp = H + (size_t)n * HID + h * OUTC;
    const float* ap = att + h * (2 * OUTC);
    float s0 = 0.f, s1 = 0.f;
#pragma unroll
    for (int c = 0; c < OUTC; ++c) {
        float hv = hp[c];
        s0 += hv * ap[c];
        s1 += hv * ap[OUTC + c];
    }
    si[t] = s0;
    sj[t] = s1;
}

// ---------------------------------------------------------------------------
// Edge scatter: one thread per (edge, head).
// alpha = leaky_relu(si[dst]+sj[src]) * invdeg[dst]; out[dst] += h[src]*alpha
// h and out are L2-resident (25.6 MB each); atomics are no-return f32 adds.
// ---------------------------------------------------------------------------
__global__ __launch_bounds__(256) void gat_edge(
        const float* __restrict__ H, const float* __restrict__ si,
        const float* __restrict__ sj, const float* __restrict__ invdeg,
        const long long* __restrict__ src, const long long* __restrict__ dst,
        float* __restrict__ out, long long nedges) {
    long long t = (long long)blockIdx.x * blockDim.x + threadIdx.x;
    if (t >= nedges * HEADS) return;
    long long e = t >> 2;
    int h = (int)(t & 3);
    int s = (int)src[e];
    int d = (int)dst[e];

    float sc = si[d * HEADS + h] + sj[s * HEADS + h];
    sc = (sc > 0.f) ? sc : NEG_SLOPE * sc;
    float alpha = sc * invdeg[d];

    const float4* hp = (const float4*)(H + (size_t)s * HID + h * OUTC);
    float* op = out + (size_t)d * HID + h * OUTC;
#pragma unroll
    for (int q = 0; q < 4; ++q) {
        float4 v = hp[q];
        unsafeAtomicAdd(op + q * 4 + 0, v.x * alpha);
        unsafeAtomicAdd(op + q * 4 + 1, v.y * alpha);
        unsafeAtomicAdd(op + q * 4 + 2, v.z * alpha);
        unsafeAtomicAdd(op + q * 4 + 3, v.w * alpha);
    }
}

// ---------------------------------------------------------------------------
// Final FC: out[n, 0:4] = x[n, 0:64] @ fcW^T + fcb
// ---------------------------------------------------------------------------
__global__ void gat_fc(const float* __restrict__ X,
                       const float* __restrict__ fcW,
                       const float* __restrict__ fcb,
                       float* __restrict__ out, int nnodes) {
    int n = blockIdx.x * blockDim.x + threadIdx.x;
    if (n >= nnodes) return;
    const float* xr = X + (size_t)n * HID;
    float a0 = fcb[0], a1 = fcb[1], a2 = fcb[2], a3 = fcb[3];
#pragma unroll 16
    for (int k = 0; k < HID; ++k) {
        float xv = xr[k];
        a0 += xv * fcW[k];
        a1 += xv * fcW[HID + k];
        a2 += xv * fcW[2 * HID + k];
        a3 += xv * fcW[3 * HID + k];
    }
    float* o = out + (size_t)n * 4;
    o[0] = a0; o[1] = a1; o[2] = a2; o[3] = a3;
}

// ---------------------------------------------------------------------------
// Launch
// ---------------------------------------------------------------------------
extern "C" void kernel_launch(void* const* d_in, const int* in_sizes, int n_in,
                              void* d_out, int out_size, void* d_ws, size_t ws_size,
                              hipStream_t stream) {
    const float* x = (const float*)d_in[0];
    const long long* ei = (const long long*)d_in[1];

    const int IN_CH = 128;
    const int N = in_sizes[0] / IN_CH;                 // 100000
    const long long E = (long long)(in_sizes[1] / 2);  // 1600000
    const long long* src = ei;
    const long long* dst = ei + E;

    const float* Wl[4] = {(const float*)d_in[2], (const float*)d_in[5],
                          (const float*)d_in[8], (const float*)d_in[11]};
    const float* Bl[4] = {(const float*)d_in[3], (const float*)d_in[6],
                          (const float*)d_in[9], (const float*)d_in[12]};
    const float* Al[4] = {(const float*)d_in[4], (const float*)d_in[7],
                          (const float*)d_in[10], (const float*)d_in[13]};
    const float* fcW = (const float*)d_in[14];
    const float* fcb = (const float*)d_in[15];

    // Workspace layout (floats): H | X1 | X2 | si | sj | deg
    float* wsf   = (float*)d_ws;
    size_t nh    = (size_t)N * HID;                    // 6.4M
    float* bufH  = wsf;
    float* bufX1 = bufH  + nh;
    float* bufX2 = bufX1 + nh;
    float* si    = bufX2 + nh;
    float* sj    = si + (size_t)N * HEADS;
    float* deg   = sj + (size_t)N * HEADS;

    const int TB = 256;

    // Degrees (same edge set for every layer), then invert once
    zero_f32<<<(N + TB - 1) / TB, TB, 0, stream>>>(deg, (size_t)N);
    gat_degree<<<(int)((E + TB - 1) / TB), TB, 0, stream>>>(dst, deg, E);
    recip_f32<<<(N + TB - 1) / TB, TB, 0, stream>>>(deg, N);

    const float* xin = x;
    int cin = IN_CH;
    float* pong[2] = {bufX1, bufX2};

    for (int l = 0; l < 4; ++l) {
        float* outb = pong[l & 1];

        int tiles  = ((N + 15) / 16) * 4;
        int gblk   = (tiles + 7) / 8;
        gat_gemm<<<gblk, TB, 0, stream>>>(xin, Wl[l], Bl[l], bufH, N, cin);

        gat_scores<<<(N * HEADS + TB - 1) / TB, TB, 0, stream>>>(bufH, Al[l], si, sj, N);

        zero_f32<<<(int)((nh + TB - 1) / TB), TB, 0, stream>>>(outb, nh);

        long long tot = E * HEADS;
        gat_edge<<<(int)((tot + TB - 1) / TB), TB, 0, stream>>>(
            bufH, si, sj, deg, src, dst, outb, E);

        xin = outb;
        cin = HID;
    }

    gat_fc<<<(N + TB - 1) / TB, TB, 0, stream>>>(xin, fcW, fcb, (float*)d_out, N);
}
